// GraphStructuralLayer_86131274154533
// MI455X (gfx1250) — matrix operand order repaired and checked by reading, not compiled, not run on hardware
//
#include <hip/hip_runtime.h>
#include <hip/hip_bf16.h>
#include <math.h>

typedef __attribute__((ext_vector_type(16))) __bf16 bf16x16;
typedef __attribute__((ext_vector_type(8)))  float  f32x8;

#define NNODES 100000
#define NEDGES 1000000
#define DIM    128
#define NHEAD  8
#define DHEAD  16
#define SCALEF 0.25f   // 1/sqrt(16)

// ---------------------------------------------------------------------------
// float atomic max via signed/unsigned int atomics (works for all finite f32
// given init to -inf): positive values compare correctly as signed ints,
// negative values compare inversely as unsigned ints.
static __device__ __forceinline__ void atomic_max_f32(float* addr, float val) {
  if (val >= 0.0f) {
    atomicMax((int*)addr, __float_as_int(val));
  } else {
    atomicMin((unsigned int*)addr, __float_as_uint(val));
  }
}

// ---------------------------------------------------------------------------
// Pack 4 f32 [128x128] weight matrices into one bf16 [128x512] + bias [512]
__global__ void pack_weights_kernel(const float* __restrict__ Wq, const float* __restrict__ bq,
                                    const float* __restrict__ Wk, const float* __restrict__ bk,
                                    const float* __restrict__ Wv, const float* __restrict__ bv,
                                    const float* __restrict__ Ws, const float* __restrict__ bs,
                                    __bf16* __restrict__ Wcat, float* __restrict__ bcat) {
  int t = blockIdx.x * blockDim.x + threadIdx.x;
  if (t >= DIM * 4 * DIM) return;
  int k  = t >> 9;        // row (input dim)
  int j  = t & 511;       // packed col
  int c  = j >> 7;        // which matrix
  int jj = j & 127;       // col within matrix
  const float* W = (c == 0) ? Wq : (c == 1) ? Wk : (c == 2) ? Wv : Ws;
  Wcat[t] = (__bf16)W[k * DIM + jj];
  if (k == 0) {
    const float* b = (c == 0) ? bq : (c == 1) ? bk : (c == 2) ? bv : bs;
    bcat[j] = b[jj];
  }
}

// f32 -> bf16 with zero padding of the tail rows (row-tile padding for GEMM)
__global__ void convert_bf16_kernel(const float* __restrict__ src, __bf16* __restrict__ dst,
                                    long nvalid, long ntotal) {
  long t = (long)blockIdx.x * blockDim.x + threadIdx.x;
  if (t >= ntotal) return;
  dst[t] = (t < nvalid) ? (__bf16)src[t] : (__bf16)0.0f;
}

__global__ void init_ms_kernel(float* __restrict__ m, float* __restrict__ s) {
  int t = blockIdx.x * blockDim.x + threadIdx.x;
  if (t >= NNODES * NHEAD) return;
  m[t] = -__builtin_inff();
  s[t] = 0.0f;
}

// ---------------------------------------------------------------------------
// Fused QKVS GEMM: Y[N,512] = Xbf[N,128] @ Wcat[128,512] + bcat
//   cols [0,384)  -> QKV buffer (row stride 384)
//   cols [384,512)-> Agg buffer (skip projection; edge atomics add on top)
// Block = 256 threads (8 wave32); block tile 128(M) x 64(N); K in 4 steps of 32.
// Wave w computes rows [w*16, w*16+16) x all 64 cols via 4 WMMA accumulators.
__global__ __launch_bounds__(256) void wmma_gemm_kernel(
    const __bf16* __restrict__ X,    // [Npad,128] bf16, zero padded
    const __bf16* __restrict__ Wc,   // [128,512] bf16
    const float*  __restrict__ bc,   // [512]
    float* __restrict__ QKV,         // [N,384]
    float* __restrict__ Agg)         // [N,128]
{
  __shared__ __bf16 Xs[128][40];     // [row][k]  (k padded 32->40)
  __shared__ __bf16 Wt[64][40];      // [n][k] transposed (k padded)

  const int t    = threadIdx.x;
  const int wave = t >> 5;
  const int lane = t & 31;
  const int half = lane >> 4;        // wave32: lanes 0-15 / 16-31
  const int nl   = lane & 15;

  const int rowBase = blockIdx.y * 128;
  const int nbase   = blockIdx.x * 64;

  f32x8 c0 = {}, c1 = {}, c2 = {}, c3 = {};

  for (int kb = 0; kb < DIM; kb += 32) {
    __syncthreads();
    // ---- stage X tile: 128 rows x 32 k (2 threads/row, 16 bf16 each)
    {
      int r   = t >> 1;
      int h16 = (t & 1) * 16;
      const uint4* g = (const uint4*)(X + (size_t)(rowBase + r) * DIM + kb + h16);
      uint4* d = (uint4*)&Xs[r][h16];
      d[0] = g[0];
      d[1] = g[1];
    }
    // ---- stage W tile transposed: 32 k x 64 n -> Wt[n][k]
    {
      int k  = t >> 3;
      int n0 = (t & 7) * 8;
      uint4 wv = *(const uint4*)(Wc + (size_t)(kb + k) * 512 + nbase + n0);
      const __bf16* wp = (const __bf16*)&wv;
#pragma unroll
      for (int i = 0; i < 8; ++i) Wt[n0 + i][k] = wp[i];
    }
    __syncthreads();

    // ---- A fragment (16-bit A 16x32 layout, ISA 7.12.2):
    // lane half selects K block: element j -> k = (j>=8)*16 + half*8 + (j&7)
    bf16x16 a;
    {
      const __bf16* xr = &Xs[wave * 16 + nl][0];
#pragma unroll
      for (int j = 0; j < 16; ++j) {
        int kk = ((j >= 8) ? 16 : 0) + half * 8 + (j & 7);
        a[j] = xr[kk];
      }
    }
    // ---- B fragments + WMMA (B 32x16: same K order as A: element j -> k = (j>=8)*16 + half*8 + (j&7))
#pragma unroll
    for (int nt = 0; nt < 4; ++nt) {
      bf16x16 b;
      const __bf16* wr = &Wt[nt * 16 + nl][0];
#pragma unroll
      for (int j = 0; j < 16; ++j) b[j] = wr[((j >= 8) ? 16 : 0) + half * 8 + (j & 7)];
      f32x8* cp = (nt == 0) ? &c0 : (nt == 1) ? &c1 : (nt == 2) ? &c2 : &c3;
      *cp = __builtin_amdgcn_wmma_f32_16x16x32_bf16(
          /*neg_a=*/false, a, /*neg_b=*/false, b,
          /*c_mod=*/(short)0, *cp, /*reuse_a=*/false, /*reuse_b=*/false);
    }
  }

  // ---- store: C/D layout — VGPR r: M = r + 8*half, N = lane&15
  const int row0 = rowBase + wave * 16 + half * 8;
#pragma unroll
  for (int nt = 0; nt < 4; ++nt) {
    const f32x8 c = (nt == 0) ? c0 : (nt == 1) ? c1 : (nt == 2) ? c2 : c3;
    const int col  = nbase + nt * 16 + nl;
    const float bias = bc[col];
#pragma unroll
    for (int r = 0; r < 8; ++r) {
      int row = row0 + r;
      if (row < NNODES) {
        float val = c[r] + bias;
        if (col < 384) QKV[(size_t)row * 384 + col] = val;             // Q|K|V
        else           Agg[(size_t)row * DIM + (col - 384)] = val;     // skip
      }
    }
  }
}

// ---------------------------------------------------------------------------
// Per (edge, head): score = <q[dst,h], k[src,h]> * SCALE ; atomic max into m[dst,h]
__global__ void edge_scores_kernel(const long long* __restrict__ ei,
                                   const float* __restrict__ QKV,
                                   float* __restrict__ scores,
                                   float* __restrict__ m) {
  int t = blockIdx.x * blockDim.x + threadIdx.x;
  if (t >= NEDGES * NHEAD) return;
  int e = t >> 3;
  int h = t & 7;
  int src = (int)ei[e];
  int dst = (int)ei[NEDGES + e];
  const float4* qp = (const float4*)(QKV + (size_t)dst * 384 + h * DHEAD);
  const float4* kp = (const float4*)(QKV + (size_t)src * 384 + DIM + h * DHEAD);
  float sum = 0.0f;
#pragma unroll
  for (int i = 0; i < 4; ++i) {
    float4 q = qp[i], k = kp[i];
    sum += q.x * k.x + q.y * k.y + q.z * k.z + q.w * k.w;
  }
  float sc = sum * SCALEF;
  scores[t] = sc;
  atomic_max_f32(&m[dst * NHEAD + h], sc);
}

// e = exp(score - m[dst,h]) ; atomic add into s[dst,h] ; store e in place
__global__ void edge_expsum_kernel(const long long* __restrict__ ei,
                                   float* __restrict__ scores,
                                   const float* __restrict__ m,
                                   float* __restrict__ s) {
  int t = blockIdx.x * blockDim.x + threadIdx.x;
  if (t >= NEDGES * NHEAD) return;
  int e = t >> 3;
  int h = t & 7;
  int dst = (int)ei[NEDGES + e];
  float ev = __expf(scores[t] - m[dst * NHEAD + h]);
  scores[t] = ev;
  atomicAdd(&s[dst * NHEAD + h], ev);
}

// Per (edge, dim): Agg[dst,d] += (e/s[dst,h]) * v[src,d]   (h = d/16)
__global__ void edge_agg_kernel(const long long* __restrict__ ei,
                                const float* __restrict__ scores,
                                const float* __restrict__ s,
                                const float* __restrict__ QKV,
                                float* __restrict__ Agg) {
  long t = (long)blockIdx.x * blockDim.x + threadIdx.x;
  if (t >= (long)NEDGES * DIM) return;
  int e = (int)(t >> 7);
  int d = (int)(t & 127);
  int h = d >> 4;
  int src = (int)ei[e];
  int dst = (int)ei[NEDGES + e];
  float alpha = scores[(size_t)e * NHEAD + h] / s[dst * NHEAD + h];
  float v = QKV[(size_t)src * 384 + 256 + d];
  atomicAdd(&Agg[(size_t)dst * DIM + d], alpha * v);
}

// out = prelu(Agg)      (Agg already contains skip + attention aggregate)
__global__ void prelu_kernel(const float* __restrict__ Agg, const float* __restrict__ a,
                             float* __restrict__ out) {
  int t = blockIdx.x * blockDim.x + threadIdx.x;
  if (t >= NNODES * DIM) return;
  float v = Agg[t];
  out[t] = (v >= 0.0f) ? v : a[0] * v;
}

// out = prelu(Agg + x)  (layer-2 residual)
__global__ void residual_prelu_kernel(const float* __restrict__ Agg, const float* __restrict__ x,
                                      const float* __restrict__ a, float* __restrict__ out) {
  int t = blockIdx.x * blockDim.x + threadIdx.x;
  if (t >= NNODES * DIM) return;
  float v = Agg[t] + x[t];
  out[t] = (v >= 0.0f) ? v : a[0] * v;
}

// ---------------------------------------------------------------------------
extern "C" void kernel_launch(void* const* d_in, const int* in_sizes, int n_in,
                              void* d_out, int out_size, void* d_ws, size_t ws_size,
                              hipStream_t stream) {
  const float*     x  = (const float*)d_in[0];
  const long long* ei = (const long long*)d_in[1];   // [2,E] int64
  const float*     a  = (const float*)d_in[2];
  const float* W1[4] = {(const float*)d_in[3],  (const float*)d_in[5],
                        (const float*)d_in[7],  (const float*)d_in[9]};
  const float* B1[4] = {(const float*)d_in[4],  (const float*)d_in[6],
                        (const float*)d_in[8],  (const float*)d_in[10]};
  const float* W2[4] = {(const float*)d_in[11], (const float*)d_in[13],
                        (const float*)d_in[15], (const float*)d_in[17]};
  const float* B2[4] = {(const float*)d_in[12], (const float*)d_in[14],
                        (const float*)d_in[16], (const float*)d_in[18]};
  float* out = (float*)d_out;

  const int Npad = ((NNODES + 127) / 128) * 128;     // 100096

  char* ws = (char*)d_ws;
  size_t off = 0;
  __bf16* xbf  = (__bf16*)(ws + off); off += (size_t)Npad * DIM * 2;
  __bf16* Wcat = (__bf16*)(ws + off); off += (size_t)DIM * 512 * 2;
  float*  bcat = (float*) (ws + off); off += 512 * 4;
  float*  QKV  = (float*) (ws + off); off += (size_t)NNODES * 384 * 4;
  float*  Agg  = (float*) (ws + off); off += (size_t)NNODES * DIM * 4;
  float*  Sc   = (float*) (ws + off); off += (size_t)NEDGES * NHEAD * 4;
  float*  Mx   = (float*) (ws + off); off += (size_t)NNODES * NHEAD * 4;
  float*  Ss   = (float*) (ws + off); off += (size_t)NNODES * NHEAD * 4;
  float*  Hb   = (float*) (ws + off); off += (size_t)NNODES * DIM * 4;

  const dim3 blk(256);
  const int gPack  = (DIM * 4 * DIM + 255) / 256;
  const int gConv  = (Npad * DIM + 255) / 256;
  const int gInit  = (NNODES * NHEAD + 255) / 256;
  const dim3 gGemm(512 / 64, Npad / 128);
  const int gEdgeH = (NEDGES * NHEAD + 255) / 256;
  const long gEdgeD = ((long)NEDGES * DIM + 255) / 256;
  const int gNode  = (NNODES * DIM + 255) / 256;

  // -------- Layer 1: input x -> Hb
  pack_weights_kernel<<<gPack, blk, 0, stream>>>(W1[0], B1[0], W1[1], B1[1],
                                                 W1[2], B1[2], W1[3], B1[3], Wcat, bcat);
  convert_bf16_kernel<<<gConv, blk, 0, stream>>>(x, xbf, (long)NNODES * DIM, (long)Npad * DIM);
  init_ms_kernel<<<gInit, blk, 0, stream>>>(Mx, Ss);
  wmma_gemm_kernel<<<gGemm, blk, 0, stream>>>(xbf, Wcat, bcat, QKV, Agg);
  edge_scores_kernel<<<gEdgeH, blk, 0, stream>>>(ei, QKV, Sc, Mx);
  edge_expsum_kernel<<<gEdgeH, blk, 0, stream>>>(ei, Sc, Mx, Ss);
  edge_agg_kernel<<<(int)gEdgeD, blk, 0, stream>>>(ei, Sc, Ss, QKV, Agg);
  prelu_kernel<<<gNode, blk, 0, stream>>>(Agg, a, Hb);

  // -------- Layer 2: input Hb -> out (with x residual)
  pack_weights_kernel<<<gPack, blk, 0, stream>>>(W2[0], B2[0], W2[1], B2[1],
                                                 W2[2], B2[2], W2[3], B2[3], Wcat, bcat);
  convert_bf16_kernel<<<gConv, blk, 0, stream>>>(Hb, xbf, (long)NNODES * DIM, (long)Npad * DIM);
  init_ms_kernel<<<gInit, blk, 0, stream>>>(Mx, Ss);
  wmma_gemm_kernel<<<gGemm, blk, 0, stream>>>(xbf, Wcat, bcat, QKV, Agg);
  edge_scores_kernel<<<gEdgeH, blk, 0, stream>>>(ei, QKV, Sc, Mx);
  edge_expsum_kernel<<<gEdgeH, blk, 0, stream>>>(ei, Sc, Mx, Ss);
  edge_agg_kernel<<<(int)gEdgeD, blk, 0, stream>>>(ei, Sc, Ss, QKV, Agg);
  residual_prelu_kernel<<<gNode, blk, 0, stream>>>(Agg, x, a, out);
}
